// MegrezMoeDecoderLayer_86320252715780
// MI455X (gfx1250) — matrix operand order repaired and checked by reading, not compiled, not run on hardware
//
#include <hip/hip_runtime.h>
#include <hip/hip_bf16.h>
#include <math.h>

// ---------------------------------------------------------------------------
// MegrezMoE decoder layer for gfx1250 (MI455X).
// bf16 WMMA everywhere; async global->LDS staging (double-buffered) feeding
// v_wmma_f32_16x16x32_bf16; weights pre-converted/transposed to bf16 so the
// GEMM hot loop is pure DMA + WMMA.
// ---------------------------------------------------------------------------

typedef __attribute__((ext_vector_type(16))) __bf16 v16bf;
typedef __attribute__((ext_vector_type(8)))  float  v8f;
typedef unsigned short ushort_t;
typedef unsigned int   uint_t;

#define DEV_INLINE __device__ __forceinline__

constexpr int   Bc   = 2;
constexpr int   Sc   = 512;
constexpr int   Dc   = 2048;
constexpr int   Hc   = 16;
constexpr int   DHc  = 128;
constexpr int   Tc   = Bc * Sc;      // 1024 tokens
constexpr int   Ec   = 8;
constexpr int   Ic   = 1024;
constexpr int   ISc  = 1024;
constexpr float EPSc = 1e-6f;
constexpr float ROUTED_SCALEc = 2.5f;

// ---- async global -> LDS (gfx1250). Guarded: falls back to sync copy. ----
#if defined(__has_builtin)
#if __has_builtin(__builtin_amdgcn_global_load_async_to_lds_b128)
#define ASYNC_LDS 1
#endif
#if __has_builtin(__builtin_amdgcn_s_wait_asynccnt)
#define WAIT_ASYNC_BUILTIN 1
#endif
#endif

// builtin prototype (from hipcc diagnostic): pointee is v4i (= int
// __attribute__((vector_size(16)))), global src in AS(1), LDS dst in AS(3).
typedef int v4i_t __attribute__((vector_size(16)));
typedef __attribute__((address_space(1))) v4i_t as1_v4i;
typedef __attribute__((address_space(3))) v4i_t as3_v4i;

DEV_INLINE void async_b128(const ushort_t* g, ushort_t* l) {
#if defined(ASYNC_LDS)
  __builtin_amdgcn_global_load_async_to_lds_b128(
      (as1_v4i*)g, (as3_v4i*)l, 0, 0);
#else
  *(uint4*)l = *(const uint4*)g;
#endif
}

template <int N> DEV_INLINE void wait_asynccnt() {
#if defined(WAIT_ASYNC_BUILTIN)
  __builtin_amdgcn_s_wait_asynccnt(N);
#else
  asm volatile("s_wait_asynccnt %0" ::"n"(N) : "memory");
#endif
}

DEV_INLINE ushort_t f2bf(float f) {
  uint_t u = __float_as_uint(f);
  u += 0x7fffu + ((u >> 16) & 1u);   // round-to-nearest-even
  return (ushort_t)(u >> 16);
}
DEV_INLINE float bf2f(ushort_t u) {
  return __uint_as_float((uint_t)u << 16);
}

union ABFrag {          // one WMMA A/B operand (16 bf16 = 8 VGPRs)
  v16bf v;
  uint4 q[2];
};

DEV_INLINE v8f wmma_bf16(v16bf a, v16bf b, v8f c) {
  return __builtin_amdgcn_wmma_f32_16x16x32_bf16(
      /*neg_a=*/false, a, /*neg_b=*/false, b,
      /*c_mod=*/(short)0, c, /*reuse_a=*/false, /*reuse_b=*/false);
}

// ---------------------------------------------------------------------------
// Weight prep: in[K,N] fp32 -> out[N,K] bf16 (transpose + convert).
// grid (N/32, K/32, batch); 256 threads.
// ---------------------------------------------------------------------------
__global__ __launch_bounds__(256) void transpose_cvt_kernel(
    const float* __restrict__ in, ushort_t* __restrict__ out, int K, int N)
{
  __shared__ float tile[32][33];
  const size_t mz = (size_t)blockIdx.z * K * N;
  const float* inz = in + mz;
  ushort_t* outz = out + mz;
  const int k0 = blockIdx.y * 32, n0 = blockIdx.x * 32;
  const int c = threadIdx.x & 31, r8 = threadIdx.x >> 5;
  #pragma unroll
  for (int i = 0; i < 4; ++i) {
    const int r = r8 + i * 8;
    tile[r][c] = inz[(size_t)(k0 + r) * N + n0 + c];
  }
  __syncthreads();
  #pragma unroll
  for (int i = 0; i < 4; ++i) {
    const int r = r8 + i * 8;                       // n-local
    outz[(size_t)(n0 + r) * K + k0 + c] = f2bf(tile[c][r]);
  }
}

// ---------------------------------------------------------------------------
// GEMM: C[M,N] (fp32) = A[M,K](bf16) @ Bt[N,K](bf16, pre-transposed).
// Epilogue: optional per-row scale then optional addsrc[m,n].
// Block 256 thr = 8 waves (2M x 4N); wave tile 32x32 (4 WMMA / k-step).
// Block tile 64(M) x 128(N), K-step 32, double-buffered async staging.
// Requires M%64==0, N%128==0, K%32==0.
// ---------------------------------------------------------------------------
__global__ __launch_bounds__(256) void gemm_bf16_async_kernel(
    const ushort_t* __restrict__ A, const ushort_t* __restrict__ Bt,
    float* __restrict__ C, int M, int N, int K,
    const float* __restrict__ addsrc, const float* __restrict__ rowscale,
    int rs_stride)
{
  (void)M;
  __shared__ __align__(16) ushort_t sA[2][64 * 32];    // 4KB x2
  __shared__ __align__(16) ushort_t sB[2][128 * 32];   // 8KB x2

  const int tid  = threadIdx.x;
  const int wave = tid >> 5, lane = tid & 31;
  const int wm   = wave >> 2, wn = wave & 3;
  const int lm   = lane & 15, lhi = lane >> 4;
  const int kb   = lhi * 8;                            // A-frag K base
  const int m0   = blockIdx.y * 64, n0 = blockIdx.x * 128;

  const int ar = tid >> 2, ac = (tid & 3) * 8;         // A: 64 rows x 32
  const int br = tid >> 1, bc = (tid & 1) * 16;        // Bt: 128 rows x 32

  v8f cc[2][2];
  #pragma unroll
  for (int i = 0; i < 2; ++i)
    #pragma unroll
    for (int j = 0; j < 2; ++j) {
      v8f z = {0, 0, 0, 0, 0, 0, 0, 0};
      cc[i][j] = z;
    }

  auto stage = [&](int k0, int buf) {
    async_b128(A + (size_t)(m0 + ar) * K + k0 + ac, &sA[buf][ar * 32 + ac]);
    const ushort_t* bp = Bt + (size_t)(n0 + br) * K + k0 + bc;
    async_b128(bp,     &sB[buf][br * 32 + bc]);
    async_b128(bp + 8, &sB[buf][br * 32 + bc + 8]);
  };

  const int nt = K >> 5;
  stage(0, 0);
  for (int t = 0; t < nt; ++t) {
    const int cur = t & 1;
    if (t + 1 < nt) { stage((t + 1) << 5, cur ^ 1); wait_asynccnt<3>(); }
    else            { wait_asynccnt<0>(); }
    __syncthreads();

    const ushort_t* pa = sA[cur];
    const ushort_t* pb = sB[cur];
    ABFrag a0, a1, b0, b1;
    const int ar0 = wm * 32 + lm, ar1 = ar0 + 16;      // A lane: m = l&15
    a0.q[0] = *(const uint4*)&pa[ar0 * 32 + kb];
    a0.q[1] = *(const uint4*)&pa[ar0 * 32 + kb + 16];
    a1.q[0] = *(const uint4*)&pa[ar1 * 32 + kb];
    a1.q[1] = *(const uint4*)&pa[ar1 * 32 + kb + 16];
    const int bn0 = wn * 32 + lm, bn1 = bn0 + 16;      // B lane: n = l&15
    b0.q[0] = *(const uint4*)&pb[bn0 * 32 + kb];
    b0.q[1] = *(const uint4*)&pb[bn0 * 32 + kb + 16];
    b1.q[0] = *(const uint4*)&pb[bn1 * 32 + kb];
    b1.q[1] = *(const uint4*)&pb[bn1 * 32 + kb + 16];

    cc[0][0] = wmma_bf16(a0.v, b0.v, cc[0][0]);
    cc[0][1] = wmma_bf16(a0.v, b1.v, cc[0][1]);
    cc[1][0] = wmma_bf16(a1.v, b0.v, cc[1][0]);
    cc[1][1] = wmma_bf16(a1.v, b1.v, cc[1][1]);
    __syncthreads();
  }

  // epilogue: C/D layout -> lane row m = r + 8*(l>>4), col n = l&15
  #pragma unroll
  for (int i = 0; i < 2; ++i)
    #pragma unroll
    for (int r = 0; r < 8; ++r) {
      const int m_l = m0 + wm * 32 + i * 16 + r + 8 * lhi;
      float vs = 1.f;
      if (rowscale) vs = rowscale[(size_t)m_l * rs_stride];
      #pragma unroll
      for (int j = 0; j < 2; ++j) {
        const int n_l = n0 + wn * 32 + j * 16 + lm;
        float v = cc[i][j][r] * vs;
        if (addsrc) v += addsrc[(size_t)m_l * N + n_l];
        C[(size_t)m_l * N + n_l] = v;
      }
    }
}

// ---------------------------------------------------------------------------
// RMSNorm over rows (fp32 in, bf16 out). One block per row.
// ---------------------------------------------------------------------------
__global__ __launch_bounds__(256) void rmsnorm_kernel(
    const float* __restrict__ x, const float* __restrict__ w,
    ushort_t* __restrict__ out, int Dn)
{
  const int row = blockIdx.x, tid = threadIdx.x;
  const float* xr = x + (size_t)row * Dn;
  float s = 0.f;
  for (int d = tid; d < Dn; d += 256) { const float v = xr[d]; s += v * v; }
  #pragma unroll
  for (int o = 16; o > 0; o >>= 1) s += __shfl_xor(s, o, 32);
  __shared__ float red[8];
  if ((tid & 31) == 0) red[tid >> 5] = s;
  __syncthreads();
  if (tid == 0) {
    float t = 0.f;
    for (int i = 0; i < 8; ++i) t += red[i];
    red[0] = rsqrtf(t / (float)Dn + EPSc);
  }
  __syncthreads();
  const float inv = red[0];
  ushort_t* orow = out + (size_t)row * Dn;
  for (int d = tid; d < Dn; d += 256) orow[d] = f2bf(xr[d] * inv * w[d]);
}

// ---------------------------------------------------------------------------
// RoPE on q,k + bf16 reorders: q,k -> [B,H,S,Dh]; v -> [B,H,Dh,S] (V^T!)
// so attention can async-DMA V^T tiles directly. grid (B*S, H), 128 thr.
// ---------------------------------------------------------------------------
__global__ __launch_bounds__(128) void rope_reorder_kernel(
    const float* __restrict__ q, const float* __restrict__ k,
    const float* __restrict__ v, ushort_t* __restrict__ qr,
    ushort_t* __restrict__ kr, ushort_t* __restrict__ vt)
{
  const int t = blockIdx.x;            // token = b*S + s
  const int h = blockIdx.y;
  const int b = t / Sc, s = t % Sc;
  const int tid = threadIdx.x;
  const size_t src  = (size_t)t * (Hc * DHc) + (size_t)h * DHc;
  const size_t dst  = ((size_t)(b * Hc + h) * Sc + s) * DHc;
  const size_t base = (size_t)(b * Hc + h) * DHc * Sc;

  vt[base + (size_t)tid * Sc + s] = f2bf(v[src + tid]);   // transposed V

  if (tid < 64) {
    const int d = tid;
    const float inv = __powf(10000.0f, -(float)d * (1.0f / 64.0f));
    const float ang = (float)s * inv;
    const float cs = __cosf(ang), sn = __sinf(ang);
    {
      const float x1 = q[src + d], x2 = q[src + 64 + d];
      qr[dst + d]      = f2bf(x1 * cs - x2 * sn);
      qr[dst + 64 + d] = f2bf(x2 * cs + x1 * sn);
    }
    {
      const float x1 = k[src + d], x2 = k[src + 64 + d];
      kr[dst + d]      = f2bf(x1 * cs - x2 * sn);
      kr[dst + 64 + d] = f2bf(x2 * cs + x1 * sn);
    }
  }
}

// ---------------------------------------------------------------------------
// Causal flash attention. grid (B*H, S/64), 128 threads = 4 waves.
// Each wave owns 16 queries; waves share async-staged K / V^T LDS tiles
// (32 keys/iter). Scores & P never leave the WGP. Output bf16 [B,S,H,Dh].
// ---------------------------------------------------------------------------
__global__ __launch_bounds__(128) void flash_attn_kernel(
    const ushort_t* __restrict__ qr, const ushort_t* __restrict__ kr,
    const ushort_t* __restrict__ vt, ushort_t* __restrict__ out)
{
  __shared__ __align__(16) ushort_t ldsK[32 * 128];   // K tile  [k][d]
  __shared__ __align__(16) ushort_t ldsVt[128 * 32];  // V^T tile [d][k]
  __shared__ __align__(16) ushort_t ldsP[4][16 * 32]; // per-wave P tile

  const int tid  = threadIdx.x;
  const int wave = tid >> 5, lane = tid & 31;
  const int lm   = lane & 15, lhi = lane >> 4;
  const int kb   = lhi * 8;
  const int bh   = blockIdx.x;
  const int b    = bh / Hc, h = bh % Hc;
  const int qblk = blockIdx.y * 64;
  const int qbase = qblk + wave * 16;
  const size_t base = (size_t)bh * Sc * DHc;   // same extent for kr and vt

  ABFrag qf[4];
  {
    const ushort_t* qp = qr + base + (size_t)(qbase + lm) * DHc;
    #pragma unroll
    for (int j = 0; j < 4; ++j) {
      qf[j].q[0] = *(const uint4*)(qp + j * 32 + kb);
      qf[j].q[1] = *(const uint4*)(qp + j * 32 + kb + 16);
    }
  }

  v8f acc[8];
  #pragma unroll
  for (int dg = 0; dg < 8; ++dg) {
    v8f z = {0, 0, 0, 0, 0, 0, 0, 0};
    acc[dg] = z;
  }
  float m_i[8], l_i[8];
  #pragma unroll
  for (int r = 0; r < 8; ++r) { m_i[r] = -__builtin_inff(); l_i[r] = 0.f; }

  const int kend = qblk + 63;
  const float scale = 0.08838834764831845f;  // 1/sqrt(128)

  for (int kt = 0; kt <= kend; kt += 32) {
    __syncthreads();
    {   // K tile [32][128]: async row chunks
      const int row = tid >> 2, c = (tid & 3) * 32;
      const ushort_t* kp = kr + base + (size_t)(kt + row) * DHc + c;
      ushort_t* lk = &ldsK[row * 128 + c];
      #pragma unroll
      for (int i = 0; i < 4; ++i) async_b128(kp + i * 8, lk + i * 8);
    }
    {   // V^T tile [128][32]: rows already transposed in global
      const ushort_t* vp = vt + base + (size_t)tid * Sc + kt;
      ushort_t* lv = &ldsVt[tid * 32];
      #pragma unroll
      for (int i = 0; i < 4; ++i) async_b128(vp + i * 8, lv + i * 8);
    }
    wait_asynccnt<0>();
    __syncthreads();
    if (kt > qbase + 15) continue;     // wave fully masked for this tile

    // ---- scores 16x32: 8 WMMAs over Dh=128 ----
    v8f s0 = {0, 0, 0, 0, 0, 0, 0, 0};
    v8f s1 = {0, 0, 0, 0, 0, 0, 0, 0};
    #pragma unroll
    for (int j = 0; j < 4; ++j) {
      ABFrag bk0, bk1;   // B[kk][n] = K[kt + half*16 + n][j*32 + kk]
      bk0.q[0] = *(const uint4*)&ldsK[lm * 128 + j * 32 + kb];
      bk0.q[1] = *(const uint4*)&ldsK[lm * 128 + j * 32 + kb + 16];
      bk1.q[0] = *(const uint4*)&ldsK[(16 + lm) * 128 + j * 32 + kb];
      bk1.q[1] = *(const uint4*)&ldsK[(16 + lm) * 128 + j * 32 + kb + 16];
      s0 = wmma_bf16(qf[j].v, bk0.v, s0);
      s1 = wmma_bf16(qf[j].v, bk1.v, s1);
    }

    // ---- online softmax (per C-layout row), P -> LDS as bf16 ----
    float corr[8];
    #pragma unroll
    for (int r = 0; r < 8; ++r) {
      const int m_g = qbase + r + 8 * lhi;
      const int n_g0 = kt + lm, n_g1 = kt + 16 + lm;
      float f0 = (n_g0 <= m_g) ? s0[r] * scale : -__builtin_inff();
      float f1 = (n_g1 <= m_g) ? s1[r] * scale : -__builtin_inff();
      float rmax = fmaxf(f0, f1);
      #pragma unroll
      for (int o = 8; o > 0; o >>= 1)
        rmax = fmaxf(rmax, __shfl_xor(rmax, o, 16));
      const float mnew = fmaxf(m_i[r], rmax);
      float cr, p0, p1;
      if (mnew == -__builtin_inff()) { cr = 1.f; p0 = 0.f; p1 = 0.f; }
      else {
        cr = __expf(m_i[r] - mnew);
        p0 = __expf(f0 - mnew);
        p1 = __expf(f1 - mnew);
      }
      float rs = p0 + p1;
      #pragma unroll
      for (int o = 8; o > 0; o >>= 1) rs += __shfl_xor(rs, o, 16);
      l_i[r] = l_i[r] * cr + rs;
      m_i[r] = mnew;
      corr[r] = cr;
      const int prow = r + 8 * lhi;
      ldsP[wave][prow * 32 + lm]      = f2bf(p0);
      ldsP[wave][prow * 32 + 16 + lm] = f2bf(p1);
    }
    #pragma unroll
    for (int dg = 0; dg < 8; ++dg)
      #pragma unroll
      for (int r = 0; r < 8; ++r) acc[dg][r] *= corr[r];

    asm volatile("" ::: "memory");     // order LDS P writes before reads

    // ---- P @ V: 8 WMMAs accumulating 16x128 output ----
    ABFrag pf;
    pf.q[0] = *(const uint4*)&ldsP[wave][lm * 32 + kb];
    pf.q[1] = *(const uint4*)&ldsP[wave][lm * 32 + kb + 16];
    #pragma unroll
    for (int dg = 0; dg < 8; ++dg) {
      ABFrag vb;                        // B[kk][n] = V[kt+kk][dg*16+n]
      vb.q[0] = *(const uint4*)&ldsVt[(dg * 16 + lm) * 32 + kb];
      vb.q[1] = *(const uint4*)&ldsVt[(dg * 16 + lm) * 32 + kb + 16];
      acc[dg] = wmma_bf16(pf.v, vb.v, acc[dg]);
    }
  }

  // ---- normalize, write bf16 out [B,S,H,Dh] ----
  #pragma unroll
  for (int r = 0; r < 8; ++r) {
    const int m_g = qbase + r + 8 * lhi;
    const float inv = 1.f / l_i[r];
    ushort_t* op = out + (((size_t)b * Sc + m_g) * Hc + h) * DHc + lm;
    #pragma unroll
    for (int dg = 0; dg < 8; ++dg) op[dg * 16] = f2bf(acc[dg][r] * inv);
  }
}

// ---------------------------------------------------------------------------
// noaux_tc grouped top-k router: dense combine weights cw[T,E] (x2.5 folded).
// x is bf16 (post-RMSNorm activations). One block per token.
// ---------------------------------------------------------------------------
__global__ __launch_bounds__(256) void route_kernel(
    const ushort_t* __restrict__ x, const float* __restrict__ gw,
    const float* __restrict__ gb, float* __restrict__ cw)
{
  const int t = blockIdx.x, tid = threadIdx.x;
  const int wave = tid >> 5, lane = tid & 31;
  float part[Ec];
  #pragma unroll
  for (int e = 0; e < Ec; ++e) part[e] = 0.f;
  const ushort_t* xr = x + (size_t)t * Dc;
  for (int d = tid; d < Dc; d += 256) {
    const float xv = bf2f(xr[d]);
    #pragma unroll
    for (int e = 0; e < Ec; ++e) part[e] += xv * gw[(size_t)e * Dc + d];
  }
  #pragma unroll
  for (int e = 0; e < Ec; ++e)
    #pragma unroll
    for (int o = 16; o > 0; o >>= 1) part[e] += __shfl_xor(part[e], o, 32);

  __shared__ float red[Ec][8];
  if (lane == 0)
    for (int e = 0; e < Ec; ++e) red[e][wave] = part[e];
  __syncthreads();
  if (tid == 0) {
    float score[Ec], sc[Ec];
    for (int e = 0; e < Ec; ++e) {
      float s = 0.f;
      for (int w = 0; w < 8; ++w) s += red[e][w];
      score[e] = 1.f / (1.f + __expf(-s));
      sc[e] = score[e] + gb[e];
    }
    float gsum[2];
    for (int g = 0; g < 2; ++g) {
      float m1 = -1e30f, m2 = -1e30f;
      for (int j = 0; j < 4; ++j) {
        const float v = sc[g * 4 + j];
        if (v > m1) { m2 = m1; m1 = v; } else if (v > m2) m2 = v;
      }
      gsum[g] = m1 + m2;
    }
    const int bg = (gsum[1] > gsum[0]) ? 1 : 0;
    int i1 = 0, i2 = 0;
    float b1v = -1e30f, b2v = -1e30f;
    for (int j = 0; j < 4; ++j) {
      const int e = bg * 4 + j;
      const float v = sc[e];
      if (v > b1v) { b2v = b1v; i2 = i1; b1v = v; i1 = e; }
      else if (v > b2v) { b2v = v; i2 = e; }
    }
    const float w1 = score[i1], w2 = score[i2];
    const float inv = 1.f / (w1 + w2);
    for (int e = 0; e < Ec; ++e) cw[(size_t)t * Ec + e] = 0.f;
    cw[(size_t)t * Ec + i1] = ROUTED_SCALEc * w1 * inv;
    cw[(size_t)t * Ec + i2] = ROUTED_SCALEc * w2 * inv;
  }
}

__global__ __launch_bounds__(256) void silu_mul_kernel(
    const float* __restrict__ g, const float* __restrict__ u,
    ushort_t* __restrict__ o, int n)
{
  const int i = blockIdx.x * 256 + threadIdx.x;
  if (i < n) {
    const float gv = g[i];
    o[i] = f2bf(gv * (1.f / (1.f + __expf(-gv))) * u[i]);
  }
}

__global__ __launch_bounds__(256) void add_kernel(
    const float* __restrict__ a, const float* __restrict__ b,
    float* __restrict__ o, int n)
{
  const int i = blockIdx.x * 256 + threadIdx.x;
  if (i < n) o[i] = a[i] + b[i];
}

// ---------------------------------------------------------------------------
// Host orchestration.
// ---------------------------------------------------------------------------
extern "C" void kernel_launch(void* const* d_in, const int* in_sizes, int n_in,
                              void* d_out, int out_size, void* d_ws,
                              size_t ws_size, hipStream_t stream)
{
  (void)in_sizes; (void)n_in; (void)out_size; (void)ws_size;

  const float* hidden  = (const float*)d_in[0];
  const float* ln1     = (const float*)d_in[1];
  const float* ln2     = (const float*)d_in[2];
  const float* wq      = (const float*)d_in[3];
  const float* wk      = (const float*)d_in[4];
  const float* wv      = (const float*)d_in[5];
  const float* wo      = (const float*)d_in[6];
  const float* gate_w  = (const float*)d_in[7];
  const float* gate_b  = (const float*)d_in[8];
  const float* w_gate  = (const float*)d_in[9];
  const float* w_up    = (const float*)d_in[10];
  const float* w_down  = (const float*)d_in[11];
  const float* sw_gate = (const float*)d_in[12];
  const float* sw_up   = (const float*)d_in[13];
  const float* sw_down = (const float*)d_in[14];
  float* out = (float*)d_out;

  char* ws = (char*)d_ws;
  const size_t MB = 1ull << 20;
  // activations
  ushort_t* xn1b  = (ushort_t*)(ws + 0 * MB);    // [T,D] bf16
  float*    q     = (float*)(ws + 4 * MB);       // [T,2048] fp32
  float*    k     = (float*)(ws + 12 * MB);
  float*    v     = (float*)(ws + 20 * MB);
  ushort_t* qr    = (ushort_t*)(ws + 28 * MB);   // [B,H,S,Dh] bf16
  ushort_t* krb   = (ushort_t*)(ws + 32 * MB);   // [B,H,S,Dh] bf16
  ushort_t* vtb   = (ushort_t*)(ws + 36 * MB);   // [B,H,Dh,S] bf16 (V^T)
  ushort_t* attnb = (ushort_t*)(ws + 40 * MB);   // [T,2048] bf16
  float*    hres  = (float*)(ws + 44 * MB);      // fp32 residual stream
  ushort_t* xn2b  = (ushort_t*)(ws + 52 * MB);   // [T,D] bf16
  float*    macc  = (float*)(ws + 56 * MB);      // MoE accumulator fp32
  float*    gbuf  = (float*)(ws + 64 * MB);      // [T,I] fp32
  float*    ubuf  = (float*)(ws + 68 * MB);
  ushort_t* abuf  = (ushort_t*)(ws + 72 * MB);   // [T,I] bf16
  float*    cw    = (float*)(ws + 74 * MB);      // [T,E]
  // pre-transposed bf16 weights (Bt = [N,K])
  ushort_t* wtq  = (ushort_t*)(ws + 75 * MB);    // [2048,2048]
  ushort_t* wtk  = (ushort_t*)(ws + 83 * MB);
  ushort_t* wtv  = (ushort_t*)(ws + 91 * MB);
  ushort_t* wto  = (ushort_t*)(ws + 99 * MB);
  ushort_t* wtg  = (ushort_t*)(ws + 107 * MB);   // E x [I,D]
  ushort_t* wtu  = (ushort_t*)(ws + 139 * MB);   // E x [I,D]
  ushort_t* wtd  = (ushort_t*)(ws + 171 * MB);   // E x [D,I]
  ushort_t* wtsg = (ushort_t*)(ws + 203 * MB);   // [IS,D]
  ushort_t* wtsu = (ushort_t*)(ws + 207 * MB);
  ushort_t* wtsd = (ushort_t*)(ws + 211 * MB);   // [D,IS]  (total ~215MB)

  const dim3 blk256(256), blk128(128);

  // ---- weight prep (transpose + bf16 convert) ----
  transpose_cvt_kernel<<<dim3(Dc / 32, Dc / 32, 1), blk256, 0, stream>>>(wq, wtq, Dc, Dc);
  transpose_cvt_kernel<<<dim3(Dc / 32, Dc / 32, 1), blk256, 0, stream>>>(wk, wtk, Dc, Dc);
  transpose_cvt_kernel<<<dim3(Dc / 32, Dc / 32, 1), blk256, 0, stream>>>(wv, wtv, Dc, Dc);
  transpose_cvt_kernel<<<dim3(Dc / 32, Dc / 32, 1), blk256, 0, stream>>>(wo, wto, Dc, Dc);
  transpose_cvt_kernel<<<dim3(Ic / 32, Dc / 32, Ec), blk256, 0, stream>>>(w_gate, wtg, Dc, Ic);
  transpose_cvt_kernel<<<dim3(Ic / 32, Dc / 32, Ec), blk256, 0, stream>>>(w_up, wtu, Dc, Ic);
  transpose_cvt_kernel<<<dim3(Dc / 32, Ic / 32, Ec), blk256, 0, stream>>>(w_down, wtd, Ic, Dc);
  transpose_cvt_kernel<<<dim3(ISc / 32, Dc / 32, 1), blk256, 0, stream>>>(sw_gate, wtsg, Dc, ISc);
  transpose_cvt_kernel<<<dim3(ISc / 32, Dc / 32, 1), blk256, 0, stream>>>(sw_up, wtsu, Dc, ISc);
  transpose_cvt_kernel<<<dim3(Dc / 32, ISc / 32, 1), blk256, 0, stream>>>(sw_down, wtsd, ISc, Dc);

  const dim3 gTD(Dc / 128, Tc / 64);   // N=2048
  const dim3 gTI(Ic / 128, Tc / 64);   // N=1024

  // ---- attention block ----
  rmsnorm_kernel<<<Tc, blk256, 0, stream>>>(hidden, ln1, xn1b, Dc);
  gemm_bf16_async_kernel<<<gTD, blk256, 0, stream>>>(xn1b, wtq, q, Tc, Dc, Dc,
                                                     nullptr, nullptr, 0);
  gemm_bf16_async_kernel<<<gTD, blk256, 0, stream>>>(xn1b, wtk, k, Tc, Dc, Dc,
                                                     nullptr, nullptr, 0);
  gemm_bf16_async_kernel<<<gTD, blk256, 0, stream>>>(xn1b, wtv, v, Tc, Dc, Dc,
                                                     nullptr, nullptr, 0);
  rope_reorder_kernel<<<dim3(Bc * Sc, Hc), blk128, 0, stream>>>(q, k, v, qr,
                                                                krb, vtb);
  flash_attn_kernel<<<dim3(Bc * Hc, Sc / 64), blk128, 0, stream>>>(qr, krb,
                                                                   vtb, attnb);
  gemm_bf16_async_kernel<<<gTD, blk256, 0, stream>>>(attnb, wto, hres, Tc, Dc,
                                                     Dc, hidden, nullptr, 0);

  // ---- MoE block ----
  rmsnorm_kernel<<<Tc, blk256, 0, stream>>>(hres, ln2, xn2b, Dc);
  route_kernel<<<Tc, blk256, 0, stream>>>(xn2b, gate_w, gate_b, cw);

  const int nTI = Tc * Ic;
  for (int e = 0; e < Ec; ++e) {
    const size_t off = (size_t)e * Dc * Ic;   // element stride per expert
    gemm_bf16_async_kernel<<<gTI, blk256, 0, stream>>>(
        xn2b, wtg + off, gbuf, Tc, Ic, Dc, nullptr, nullptr, 0);
    gemm_bf16_async_kernel<<<gTI, blk256, 0, stream>>>(
        xn2b, wtu + off, ubuf, Tc, Ic, Dc, nullptr, nullptr, 0);
    silu_mul_kernel<<<(nTI + 255) / 256, blk256, 0, stream>>>(gbuf, ubuf, abuf,
                                                              nTI);
    gemm_bf16_async_kernel<<<gTD, blk256, 0, stream>>>(
        abuf, wtd + off, macc, Tc, Dc, Ic,
        (e == 0) ? nullptr : macc, cw + e, Ec);
  }

  // ---- shared expert ----
  gemm_bf16_async_kernel<<<gTI, blk256, 0, stream>>>(xn2b, wtsg, gbuf, Tc, ISc,
                                                     Dc, nullptr, nullptr, 0);
  gemm_bf16_async_kernel<<<gTI, blk256, 0, stream>>>(xn2b, wtsu, ubuf, Tc, ISc,
                                                     Dc, nullptr, nullptr, 0);
  silu_mul_kernel<<<(Tc * ISc + 255) / 256, blk256, 0, stream>>>(gbuf, ubuf,
                                                                 abuf,
                                                                 Tc * ISc);
  gemm_bf16_async_kernel<<<gTD, blk256, 0, stream>>>(abuf, wtsd, macc, Tc, Dc,
                                                     ISc, macc, nullptr, 0);

  // ---- final residual ----
  add_kernel<<<(Tc * Dc + 255) / 256, blk256, 0, stream>>>(hres, macc, out,
                                                           Tc * Dc);
}
